// SimMLP_21534966022864
// MI455X (gfx1250) — compile-verified, matrix operand order checked by reading, not run on hardware
//
#include <hip/hip_runtime.h>
#include <cstddef>
#include <cstdint>

// ---------------- problem constants ----------------
#define BB     8
#define NN     4096
#define CIN    64
#define COUT   128
#define NPOINT 1024
#define KNN    24
#define KPAD   32
#define MGRP   (BB * NPOINT)          // 8192 groups
#define RLOC   (MGRP * KPAD)          // 262144 rows, local-conv GEMM
#define RCA    (BB * NPOINT)          // 8192 rows, ca GEMM

typedef __attribute__((ext_vector_type(16))) __bf16 v16bf;
typedef __attribute__((ext_vector_type(8)))  float  v8f;

union Q16 { uint4 q; __bf16 h[8]; };
union A32 { v16bf v; uint4 q[2]; };

// =====================================================================
// 1) Weight conversion f32 -> bf16 into one packed buffer
// =====================================================================
__global__ __launch_bounds__(256)
void wprep_kernel(const float* __restrict__ wproj, const float* __restrict__ wla1,
                  const float* __restrict__ wla2,  const float* __restrict__ wca1,
                  const float* __restrict__ wca2,  __bf16* __restrict__ Wbf)
{
    int t = blockIdx.x * 256 + threadIdx.x;
    const int NPJ = 128 * 64, NL = 128 * 128;
    if (t < NPJ)                Wbf[t] = (__bf16)wproj[t];
    else if (t < NPJ + NL)      Wbf[t] = (__bf16)wla1[t - NPJ];
    else if (t < NPJ + 2 * NL)  Wbf[t] = (__bf16)wla2[t - NPJ - NL];
    else if (t < NPJ + 3 * NL)  Wbf[t] = (__bf16)wca1[t - NPJ - 2 * NL];
    else if (t < NPJ + 4 * NL)  Wbf[t] = (__bf16)wca2[t - NPJ - 3 * NL];
}

// =====================================================================
// 2) Farthest point sampling. One block per batch, xyz resident in LDS,
//    per-thread min-dist in registers, wave32 + block argmax.
// =====================================================================
__global__ __launch_bounds__(256)
void fps_kernel(const float* __restrict__ xyz, const int* __restrict__ far0,
                int* __restrict__ fps_idx)
{
    __shared__ float px[NN], py[NN], pz[NN];
    __shared__ float rmax[8];
    __shared__ int   rid[8];
    __shared__ int   scur;

    const int b = blockIdx.x, tid = threadIdx.x;
    const float* src = xyz + (size_t)b * NN * 3;
    for (int i = tid; i < NN; i += 256) {
        px[i] = src[i * 3 + 0];
        py[i] = src[i * 3 + 1];
        pz[i] = src[i * 3 + 2];
    }
    float md[16];
#pragma unroll
    for (int j = 0; j < 16; ++j) md[j] = 1e10f;
    if (tid == 0) scur = far0[b];
    __syncthreads();

    for (int s = 0; s < NPOINT; ++s) {
        const int cur = scur;
        if (tid == 0) fps_idx[b * NPOINT + s] = cur;
        const float cx = px[cur], cy = py[cur], cz = pz[cur];
        float bmax = -1.0f; int bidx = 0;
#pragma unroll
        for (int j = 0; j < 16; ++j) {
            const int i = tid + j * 256;
            const float dx = px[i] - cx, dy = py[i] - cy, dz = pz[i] - cz;
            float d = fminf(md[j], dx * dx + dy * dy + dz * dz);
            md[j] = d;
            if (d > bmax) { bmax = d; bidx = i; }
        }
        for (int off = 16; off; off >>= 1) {
            float om = __shfl_xor(bmax, off);
            int   oi = __shfl_xor(bidx, off);
            if (om > bmax || (om == bmax && oi < bidx)) { bmax = om; bidx = oi; }
        }
        if ((tid & 31) == 0) { rmax[tid >> 5] = bmax; rid[tid >> 5] = bidx; }
        __syncthreads();
        if (tid == 0) {
            float m = rmax[0]; int id = rid[0];
            for (int i = 1; i < 8; ++i)
                if (rmax[i] > m || (rmax[i] == m && rid[i] < id)) { m = rmax[i]; id = rid[i]; }
            scur = id;
        }
        __syncthreads();
    }
}

// =====================================================================
// 3) new_xyz gather -> first 24576 floats of d_out
// =====================================================================
__global__ __launch_bounds__(256)
void newxyz_kernel(const float* __restrict__ xyz, const int* __restrict__ fps_idx,
                   float* __restrict__ out)
{
    int t = blockIdx.x * 256 + threadIdx.x;
    if (t >= BB * NPOINT * 3) return;
    const int j = t % 3, g = t / 3, b = g >> 10;
    const int idx = fps_idx[g];
    out[t] = xyz[((size_t)b * NN + idx) * 3 + j];
}

// =====================================================================
// 4) kNN: one block per query, distances in LDS, 24 cooperative arg-mins
// =====================================================================
__global__ __launch_bounds__(256)
void knn_kernel(const float* __restrict__ xyz, const int* __restrict__ fps_idx,
                int* __restrict__ knn)
{
    __shared__ float dist[NN];
    __shared__ float rmin[8];
    __shared__ int   rid[8];

    const int g = blockIdx.x;            // b*NPOINT + p
    const int b = g >> 10;
    const int tid = threadIdx.x;
    const float* src = xyz + (size_t)b * NN * 3;
    const int ci = fps_idx[g];
    const float cx = src[ci * 3], cy = src[ci * 3 + 1], cz = src[ci * 3 + 2];

    for (int i = tid; i < NN; i += 256) {
        const float dx = src[i * 3] - cx, dy = src[i * 3 + 1] - cy, dz = src[i * 3 + 2] - cz;
        dist[i] = dx * dx + dy * dy + dz * dz;
    }
    __syncthreads();

    for (int k = 0; k < KNN; ++k) {
        float bmin = 3.4e38f; int bidx = 0;
        for (int i = tid; i < NN; i += 256) {
            const float d = dist[i];
            if (d < bmin || (d == bmin && i < bidx)) { bmin = d; bidx = i; }
        }
        for (int off = 16; off; off >>= 1) {
            float om = __shfl_xor(bmin, off);
            int   oi = __shfl_xor(bidx, off);
            if (om < bmin || (om == bmin && oi < bidx)) { bmin = om; bidx = oi; }
        }
        if ((tid & 31) == 0) { rmin[tid >> 5] = bmin; rid[tid >> 5] = bidx; }
        __syncthreads();
        if (tid == 0) {
            float m = rmin[0]; int id = rid[0];
            for (int i = 1; i < 8; ++i)
                if (rmin[i] < m || (rmin[i] == m && rid[i] < id)) { m = rmin[i]; id = rid[i]; }
            knn[g * KNN + k] = id;
            dist[id] = 3.4e38f;            // remove from candidate set
        }
        __syncthreads();
    }
}

// =====================================================================
// 5) Edge features, bf16, k-major layout E[m][k(0..31)][c(0..63)], k>=24 -> 0
// =====================================================================
__global__ __launch_bounds__(256)
void edge_kernel(const float* __restrict__ x, const int* __restrict__ fps_idx,
                 const int* __restrict__ knn, __bf16* __restrict__ E)
{
    const int m = blockIdx.x, b = m >> 10;
    const int tid = threadIdx.x;
    const int c = tid & 63;
    const int kb = tid >> 6;                 // 0..3
    const float* xb = x + (size_t)b * CIN * NN + (size_t)c * NN;
    const float cv = xb[fps_idx[m]];
    __bf16* Eg = E + (size_t)m * KPAD * CIN;
    for (int k = kb; k < KPAD; k += 4) {
        float v = 0.0f;
        if (k < KNN) v = xb[knn[m * KNN + k]] - cv;
        Eg[k * CIN + c] = (__bf16)v;
    }
}

// =====================================================================
// 6) GEMM + fused BN-stats: Z[r, o] = sum_c act(X[r, c]) * W[o, c]
//    act = identity, or fused BN+ReLU of previous stage (affine != null).
//    - CinT templated (64/128) -> fully unrolled k-chain, all fragments
//      loaded in one clause before the WMMA chain.
//    - 32 rows per block: A fragment reused by two B fragments, two
//      independent accumulators -> back-to-back independent v_wmma.
//    - per-channel sum / sum^2 reduced from the f32 accumulators
//      (block LDS reduction + global atomics); KMASK: only (r&31)<24 rows.
// =====================================================================
template<int CinT, int KMASK>
__global__ __launch_bounds__(256)
void gemm_kernel(const __bf16* __restrict__ X, const __bf16* __restrict__ W,
                 __bf16* __restrict__ Z,
                 const float* __restrict__ affine,  // [a[128], bias[128]] or null
                 float* __restrict__ sums)          // [sum[128], sq[128]]
{
    constexpr int NK = CinT / 32;
    __shared__ __bf16 xt[32 * CinT];          // 32 rows x Cin channels (<= 8 KB)
    __shared__ float  bsum[COUT], bsq[COUT];

    const int r0  = blockIdx.x * 32;
    const int tid = threadIdx.x;
    if (tid < COUT) { bsum[tid] = 0.0f; bsq[tid] = 0.0f; }

    // ---- stage 32 X rows into LDS (optionally with fused BN+ReLU) ----
    constexpr int ELEMS = 32 * CinT;
#pragma unroll
    for (int base = 0; base < ELEMS; base += 256 * 8) {
        const int e   = base + tid * 8;
        const int row = e / CinT;
        const int col = e % CinT;
        Q16 u;
        u.q = *(const uint4*)(X + (size_t)(r0 + row) * CinT + col);
        __builtin_prefetch(X + (size_t)(r0 + 32 + row) * CinT + col, 0, 1);
        if (affine) {
#pragma unroll
            for (int j = 0; j < 8; ++j) {
                float v = (float)u.h[j];
                v = fmaxf(affine[col + j] * v + affine[128 + col + j], 0.0f);
                u.h[j] = (__bf16)v;
            }
        }
        *(uint4*)(xt + row * CinT + col) = u.q;
    }
    __syncthreads();

    const int wave = tid >> 5;
    const int lane = tid & 31;
    const int o0   = wave * 16;
    const int frow = lane & 15;
    const int off  = (lane >> 4) * 8;

    // ---- load ALL fragments first (one clause), then the WMMA chain ----
    A32 a[NK], b0[NK], b1[NK];
#pragma unroll
    for (int s = 0; s < NK; ++s) {
        const __bf16* wrow = W + (size_t)(o0 + frow) * CinT + s * 32 + off;
        a[s].q[0] = *(const uint4*)(wrow);
        a[s].q[1] = *(const uint4*)(wrow + 16);
        const __bf16* xr0 = xt + frow * CinT + s * 32 + off;          // ds_load_b128
        b0[s].q[0] = *(const uint4*)(xr0);
        b0[s].q[1] = *(const uint4*)(xr0 + 16);
        const __bf16* xr1 = xt + (16 + frow) * CinT + s * 32 + off;
        b1[s].q[0] = *(const uint4*)(xr1);
        b1[s].q[1] = *(const uint4*)(xr1 + 16);
    }
    v8f acc0 = {}, acc1 = {};
#pragma unroll
    for (int s = 0; s < NK; ++s) {
        acc0 = __builtin_amdgcn_wmma_f32_16x16x32_bf16(false, a[s].v, false, b0[s].v,
                                                       (short)0, acc0, false, false);
        acc1 = __builtin_amdgcn_wmma_f32_16x16x32_bf16(false, a[s].v, false, b1[s].v,
                                                       (short)0, acc1, false, false);
    }

    // ---- store D fragments (bf16) ----
    Q16 q0, q1;
#pragma unroll
    for (int j = 0; j < 8; ++j) { q0.h[j] = (__bf16)acc0[j]; q1.h[j] = (__bf16)acc1[j]; }
    *(uint4*)(Z + (size_t)(r0 + frow)      * COUT + o0 + off) = q0.q;
    *(uint4*)(Z + (size_t)(r0 + 16 + frow) * COUT + o0 + off) = q1.q;

    // ---- fused per-channel batch stats from the f32 accumulators ----
    // acc0 columns: (r&31) = frow       -> always valid (frow < 24)
    // acc1 columns: (r&31) = 16 + frow  -> valid iff frow < 8 (when KMASK)
    const bool v1 = (!KMASK) || (frow < 8);
#pragma unroll
    for (int j = 0; j < 8; ++j) {
        const float x0 = acc0[j];
        const float x1 = v1 ? acc1[j] : 0.0f;
        atomicAdd(&bsum[o0 + off + j], x0 + x1);
        atomicAdd(&bsq [o0 + off + j], x0 * x0 + x1 * x1);
    }
    __syncthreads();
    if (tid < COUT) {
        atomicAdd(&sums[tid],        bsum[tid]);
        atomicAdd(&sums[COUT + tid], bsq[tid]);
    }
}

// =====================================================================
// 7) Stats -> affine: a = g*rsqrt(var+eps); bias = b - a*mean
// =====================================================================
__global__ void finalize_kernel(const float* __restrict__ sums,
                                const float* __restrict__ g,
                                const float* __restrict__ bvec,
                                float inv_count, float* __restrict__ affine)
{
    const int c = threadIdx.x;
    if (c >= COUT) return;
    const float mean = sums[c] * inv_count;
    const float var  = sums[COUT + c] * inv_count - mean * mean;
    const float a    = g[c] * rsqrtf(var + 1e-5f);
    affine[c]        = a;
    affine[COUT + c] = bvec[c] - a * mean;
}

// =====================================================================
// 8) Local epilogue: h = relu(bn3(z3) + relu(bn1(z1))), max-pool over k<24
// =====================================================================
__global__ __launch_bounds__(128)
void pool_kernel(const __bf16* __restrict__ z3, const __bf16* __restrict__ z1,
                 const float* __restrict__ aff3, const float* __restrict__ aff1,
                 __bf16* __restrict__ H)
{
    const int m = blockIdx.x, o = threadIdx.x;
    const float a3 = aff3[o], b3 = aff3[COUT + o];
    const float a1 = aff1[o], b1 = aff1[COUT + o];
    float best = -3.4e38f;
    for (int k = 0; k < KNN; ++k) {
        const size_t r = (size_t)(m * KPAD + k) * COUT + o;
        const float h1 = fmaxf(a1 * (float)z1[r] + b1, 0.0f);
        const float v  = fmaxf(a3 * (float)z3[r] + b3 + h1, 0.0f);
        best = fmaxf(best, v);
    }
    H[(size_t)m * COUT + o] = (__bf16)best;
}

// =====================================================================
// 9) ca epilogue: out[b,o,p] = relu(bn5(z5) + H)  (transpose scatter)
// =====================================================================
__global__ __launch_bounds__(128)
void cafinal_kernel(const __bf16* __restrict__ z5, const __bf16* __restrict__ H,
                    const float* __restrict__ aff5, float* __restrict__ outh)
{
    const int r = blockIdx.x;            // b*NPOINT + p
    const int o = threadIdx.x;
    const int b = r >> 10, p = r & 1023;
    const float v = fmaxf(aff5[o] * (float)z5[(size_t)r * COUT + o] + aff5[COUT + o]
                          + (float)H[(size_t)r * COUT + o], 0.0f);
    outh[((size_t)(b * COUT) + o) * NPOINT + p] = v;
}

// =====================================================================
// host: kernel_launch
// =====================================================================
extern "C" void kernel_launch(void* const* d_in, const int* in_sizes, int n_in,
                              void* d_out, int out_size, void* d_ws, size_t ws_size,
                              hipStream_t stream)
{
    (void)in_sizes; (void)n_in; (void)out_size; (void)ws_size;
    const float* xyz    = (const float*)d_in[0];
    const float* x      = (const float*)d_in[1];
    const int*   far0   = (const int*)  d_in[2];
    const float* Wproj  = (const float*)d_in[3];
    const float* g_proj = (const float*)d_in[4];
    const float* b_proj = (const float*)d_in[5];
    const float* Wla1   = (const float*)d_in[6];
    const float* g_la1  = (const float*)d_in[7];
    const float* b_la1  = (const float*)d_in[8];
    const float* Wla2   = (const float*)d_in[9];
    const float* g_la2  = (const float*)d_in[10];
    const float* b_la2  = (const float*)d_in[11];
    const float* Wca1   = (const float*)d_in[12];
    const float* g_ca1  = (const float*)d_in[13];
    const float* b_ca1  = (const float*)d_in[14];
    const float* Wca2   = (const float*)d_in[15];
    const float* g_ca2  = (const float*)d_in[16];
    const float* b_ca2  = (const float*)d_in[17];

    float* out_xyz = (float*)d_out;                    // [B, NPOINT, 3]
    float* out_h   = (float*)d_out + BB * NPOINT * 3;  // [B, COUT, NPOINT]

    // ---- workspace carve-up (256B aligned) ----
    char* base = (char*)d_ws;
    size_t off = 0;
    auto carve = [&](size_t bytes) {
        char* p = base + off;
        off = (off + bytes + 255) & ~(size_t)255;
        return p;
    };
    int*    fps_idx = (int*)   carve((size_t)MGRP * 4);
    int*    knn     = (int*)   carve((size_t)MGRP * KNN * 4);
    float*  stats   = (float*) carve(5 * 256 * 4);          // 5 stages x [sum,sq]x128
    float*  aff     = (float*) carve(5 * 256 * 4);          // 5 stages x [a,bias]x128
    __bf16* Wbf     = (__bf16*)carve((size_t)(128 * 64 + 4 * 128 * 128) * 2);
    __bf16* E       = (__bf16*)carve((size_t)MGRP * KPAD * CIN * 2);
    __bf16* z1      = (__bf16*)carve((size_t)RLOC * COUT * 2);
    __bf16* z2      = (__bf16*)carve((size_t)RLOC * COUT * 2);
    __bf16* z3      = (__bf16*)carve((size_t)RLOC * COUT * 2);
    __bf16* H       = (__bf16*)carve((size_t)RCA * COUT * 2);
    __bf16* z4      = (__bf16*)carve((size_t)RCA * COUT * 2);
    __bf16* z5      = (__bf16*)carve((size_t)RCA * COUT * 2);

    __bf16* Wp_bf  = Wbf;
    __bf16* Wl1_bf = Wbf + 128 * 64;
    __bf16* Wl2_bf = Wl1_bf + 128 * 128;
    __bf16* Wc1_bf = Wl2_bf + 128 * 128;
    __bf16* Wc2_bf = Wc1_bf + 128 * 128;

    hipMemsetAsync(stats, 0, 5 * 256 * 4, stream);

    // weights -> bf16
    wprep_kernel<<<(128 * 64 + 4 * 128 * 128 + 255) / 256, 256, 0, stream>>>(
        Wproj, Wla1, Wla2, Wca1, Wca2, Wbf);

    // FPS, new_xyz, kNN, edge features
    fps_kernel<<<BB, 256, 0, stream>>>(xyz, far0, fps_idx);
    newxyz_kernel<<<(BB * NPOINT * 3 + 255) / 256, 256, 0, stream>>>(xyz, fps_idx, out_xyz);
    knn_kernel<<<MGRP, 256, 0, stream>>>(xyz, fps_idx, knn);
    edge_kernel<<<MGRP, 256, 0, stream>>>(x, fps_idx, knn, E);

    const float invLoc = 1.0f / (float)(MGRP * KNN);
    const float invCa  = 1.0f / (float)RCA;

    // Local stage: conv (+fused stats) -> affine -> next conv (BN+ReLU fused on load)
    gemm_kernel<CIN, 1><<<RLOC / 32, 256, 0, stream>>>(E, Wp_bf, z1, nullptr, stats + 0 * 256);
    finalize_kernel<<<1, 128, 0, stream>>>(stats + 0 * 256, g_proj, b_proj, invLoc, aff + 0 * 256);

    gemm_kernel<COUT, 1><<<RLOC / 32, 256, 0, stream>>>(z1, Wl1_bf, z2, aff + 0 * 256, stats + 1 * 256);
    finalize_kernel<<<1, 128, 0, stream>>>(stats + 1 * 256, g_la1, b_la1, invLoc, aff + 1 * 256);

    gemm_kernel<COUT, 1><<<RLOC / 32, 256, 0, stream>>>(z2, Wl2_bf, z3, aff + 1 * 256, stats + 2 * 256);
    finalize_kernel<<<1, 128, 0, stream>>>(stats + 2 * 256, g_la2, b_la2, invLoc, aff + 2 * 256);

    // residual + relu + max-pool over K -> H [RCA, COUT]
    pool_kernel<<<MGRP, 128, 0, stream>>>(z3, z1, aff + 2 * 256, aff + 0 * 256, H);

    // ca resblock
    gemm_kernel<COUT, 0><<<RCA / 32, 256, 0, stream>>>(H, Wc1_bf, z4, nullptr, stats + 3 * 256);
    finalize_kernel<<<1, 128, 0, stream>>>(stats + 3 * 256, g_ca1, b_ca1, invCa, aff + 3 * 256);

    gemm_kernel<COUT, 0><<<RCA / 32, 256, 0, stream>>>(z4, Wc2_bf, z5, aff + 3 * 256, stats + 4 * 256);
    finalize_kernel<<<1, 128, 0, stream>>>(stats + 4 * 256, g_ca2, b_ca2, invCa, aff + 4 * 256);

    cafinal_kernel<<<RCA, 128, 0, stream>>>(z5, H, aff + 4 * 256, out_h);
}
